// T5Attention_81131932221777
// MI455X (gfx1250) — matrix operand order repaired and checked by reading, not compiled, not run on hardware
//
#include <hip/hip_runtime.h>

typedef _Float16 half_t;
typedef __attribute__((ext_vector_type(16))) _Float16 v16h;
typedef __attribute__((ext_vector_type(8)))  _Float16 v8h;
typedef __attribute__((ext_vector_type(8)))  float    v8f;

// exact parameter types for the async-LDS builtin (per hipcc diagnostic:
// "int __vector(4) __device__ *")
typedef int v4i __attribute__((vector_size(16)));
typedef v4i __attribute__((address_space(1))) v4i_g;
typedef v4i __attribute__((address_space(3))) v4i_l;

#define HIDDEN 512
#define NHEADS 8
#define HEADD  64
#define BATCH  4
#define SEQ    2048
#define ROWS   (BATCH * SEQ)  /* 8192 */
#define NTBLK  (SEQ / 64)     /* 32 key blocks per row-tile */

#if defined(__has_builtin)
#if __has_builtin(__builtin_amdgcn_global_load_async_to_lds_b128)
#define USE_ASYNC_LDS 1
#endif
#endif

// ---------------------------------------------------------------------------
// WMMA helpers (CDNA5 gfx1250, wave32)
// ---------------------------------------------------------------------------
static __device__ __forceinline__ v8f wmma_f32_f16(v16h a, v16h b, v8f c) {
  // D = A(16x32 f16) * B(32x16 f16) + C(16x16 f32)
  return __builtin_amdgcn_wmma_f32_16x16x32_f16(false, a, false, b, (short)0, c,
                                                false, false);
}

// A-fragment: 16x32 (MxK), source row-major [row][K] (ISA 7.12.2):
// lane L: M = L%16, g = L/16; halves 0..7 = mem[M][k0+g*8 .. +7],
//                             halves 8..15 = mem[M][k0+16+g*8 .. +7]
static __device__ __forceinline__ v16h load_a_frag(const half_t* p, int ld,
                                                   int row_base, int k0) {
  const int lane = threadIdx.x & 31;
  const int m = row_base + (lane & 15);
  const int g = lane >> 4;
  const half_t* q = p + m * ld + k0 + g * 8;
  v8h lo = *(const v8h*)(q);
  v8h hi = *(const v8h*)(q + 16);
  return __builtin_shufflevector(lo, hi, 0, 1, 2, 3, 4, 5, 6, 7, 8, 9, 10, 11,
                                 12, 13, 14, 15);
}

// B-fragment: 32x16 (KxN), source stored column-major as [n][K] row-major.
// lane L: N = L%16, g = L/16; halves 0..7 = K g*8..+7, halves 8..15 = K 16+g*8..+7.
static __device__ __forceinline__ v16h load_b_frag(const half_t* p, int ld,
                                                   int col_base, int k0) {
  const int lane = threadIdx.x & 31;
  const int n = col_base + (lane & 15);
  const int g = lane >> 4;
  const half_t* q = p + n * ld + k0 + g * 8;
  v8h lo = *(const v8h*)(q);
  v8h hi = *(const v8h*)(q + 16);
  return __builtin_shufflevector(lo, hi, 0, 1, 2, 3, 4, 5, 6, 7, 8, 9, 10, 11,
                                 12, 13, 14, 15);
}

// ---------------------------------------------------------------------------
// Async global->LDS staging (gfx1250 GLOBAL_LOAD_ASYNC_TO_LDS_B128, ASYNCcnt)
// with a synchronous fallback when the builtin is unavailable.
// ---------------------------------------------------------------------------
static __device__ __forceinline__ void copy16_g2l(const half_t* gsrc,
                                                  half_t* ldst) {
#if USE_ASYNC_LDS
  __builtin_amdgcn_global_load_async_to_lds_b128((v4i_g*)gsrc, (v4i_l*)ldst, 0,
                                                 0);
#else
  *(v8h*)ldst = *(const v8h*)gsrc;
#endif
}

static __device__ __forceinline__ void wait_async_le4() {
#if USE_ASYNC_LDS
  asm volatile("s_wait_asynccnt 4" ::: "memory");
#endif
}
static __device__ __forceinline__ void wait_async_0() {
#if USE_ASYNC_LDS
  asm volatile("s_wait_asynccnt 0" ::: "memory");
#endif
}

// ---------------------------------------------------------------------------
// Conversion kernels
// ---------------------------------------------------------------------------
__global__ void k_convert_x(const float* __restrict__ x, half_t* __restrict__ xh,
                            int n) {
  for (int i = blockIdx.x * blockDim.x + threadIdx.x; i < n;
       i += gridDim.x * blockDim.x)
    xh[i] = (half_t)x[i];
}

// W [k][n] f32 -> Wt [n][k] f16 (transpose for contiguous B-frag loads).
// Wq additionally scaled by 1/sqrt(HEAD_DIM) = 0.125 (exact in f16).
__global__ void k_convert_w(const float* __restrict__ Wq,
                            const float* __restrict__ Wk,
                            const float* __restrict__ Wv,
                            const float* __restrict__ Wo,
                            half_t* __restrict__ Wqt, half_t* __restrict__ Wkt,
                            half_t* __restrict__ Wvt, half_t* __restrict__ Wot) {
  const float* src;
  half_t* dst;
  float sc = 1.0f;
  switch (blockIdx.y) {
    case 0:  src = Wq; dst = Wqt; sc = 0.125f; break;
    case 1:  src = Wk; dst = Wkt; break;
    case 2:  src = Wv; dst = Wvt; break;
    default: src = Wo; dst = Wot; break;
  }
  for (int i = blockIdx.x * blockDim.x + threadIdx.x; i < HIDDEN * HIDDEN;
       i += gridDim.x * blockDim.x) {
    int kk = i >> 9, n = i & 511;  // i = kk*512 + n
    dst[n * HIDDEN + kk] = (half_t)(src[i] * sc);
  }
}

// Expand bias_table[32][8] to bias_full[h][rel+2048] in f16, rel = t - s.
__global__ void k_bias(const float* __restrict__ bias_table,
                       half_t* __restrict__ bias_full) {
  int i = blockIdx.x * blockDim.x + threadIdx.x;
  if (i >= 4096) return;
  int rel = i - 2048;
  int ret = rel > 0 ? 16 : 0;
  int a = rel < 0 ? -rel : rel;
  int bucket;
  if (a < 8) {
    bucket = a;
  } else {
    // 8 + log(a/8)/log(16) * 8, truncated; clamp to 15
    int v = 8 + (int)(__logf((float)a * 0.125f) * 2.88539008f);
    bucket = v < 15 ? v : 15;
  }
  bucket += ret;
#pragma unroll
  for (int h = 0; h < NHEADS; ++h)
    bias_full[h * 4096 + i] = (half_t)bias_table[bucket * NHEADS + h];
}

// ---------------------------------------------------------------------------
// GEMM core: 64-col weight strip staged in LDS (shared by all 8 waves),
// two 256-deep k-phases; A streamed from global with register double-buffer.
// ---------------------------------------------------------------------------
template <typename EPILOG>
static __device__ __forceinline__ void gemm_strip(const half_t* __restrict__ asrc,
                                                  const half_t* __restrict__ wt,
                                                  half_t* wls, int m0, int n0,
                                                  EPILOG epi) {
  const int tid = threadIdx.x;
  v8f c0 = {}, c1 = {}, c2 = {}, c3 = {};
  for (int ph = 0; ph < 2; ++ph) {
    if (ph) __syncthreads();  // phase-0 reads done before overwrite
    // cooperative stage: 64 rows x 256 halves = 32 KB = 2048 x 16B chunks
#pragma unroll
    for (int j = 0; j < 8; ++j) {
      int c = tid + j * 256;
      int n = c >> 5, o = (c & 31) * 8;
      copy16_g2l(wt + (n0 + n) * HIDDEN + ph * 256 + o, wls + n * 256 + o);
    }
    wait_async_0();
    __syncthreads();

    const int kb = ph * 256;
    v16h a  = load_a_frag(asrc, HIDDEN, m0, kb);
    v16h b0 = load_b_frag(wls, 256, 0, 0);
    v16h b1 = load_b_frag(wls, 256, 16, 0);
    v16h b2 = load_b_frag(wls, 256, 32, 0);
    v16h b3 = load_b_frag(wls, 256, 48, 0);
#pragma unroll
    for (int kl = 0; kl < 256; kl += 32) {
      const int kn = kl + 32;
      const int kp = (kn < 256) ? kn : kl;  // last iter: dummy reload
      v16h an  = load_a_frag(asrc, HIDDEN, m0, kb + kp);
      v16h bn0 = load_b_frag(wls, 256, 0, kp);
      v16h bn1 = load_b_frag(wls, 256, 16, kp);
      v16h bn2 = load_b_frag(wls, 256, 32, kp);
      v16h bn3 = load_b_frag(wls, 256, 48, kp);
      c0 = wmma_f32_f16(a, b0, c0);
      c1 = wmma_f32_f16(a, b1, c1);
      c2 = wmma_f32_f16(a, b2, c2);
      c3 = wmma_f32_f16(a, b3, c3);
      a = an; b0 = bn0; b1 = bn1; b2 = bn2; b3 = bn3;
    }
  }
  epi(c0, c1, c2, c3);
}

// QKV projection: grid (64, 8, 3), block 256; Q,K head-major, V transposed.
__global__ void __launch_bounds__(256) k_gemm_qkv(
    const half_t* __restrict__ xh, const half_t* __restrict__ Wqt,
    const half_t* __restrict__ Wkt, const half_t* __restrict__ Wvt,
    half_t* __restrict__ Qh, half_t* __restrict__ Kh, half_t* __restrict__ Vt) {
  __shared__ __align__(16) half_t wls[64 * 256];  // 32 KB
  const int wave = threadIdx.x >> 5;
  const int lane = threadIdx.x & 31;
  const int m0 = blockIdx.x * 128 + wave * 16;
  const int n0 = blockIdx.y * 64;
  const int mat = blockIdx.z;
  const half_t* wt = (mat == 0) ? Wqt : (mat == 1 ? Wkt : Wvt);

  gemm_strip(xh, wt, wls, m0, n0, [&](v8f c0, v8f c1, v8f c2, v8f c3) {
    const int g = lane >> 4;
    const int nl = lane & 15;
    v8f cc[4] = {c0, c1, c2, c3};
#pragma unroll
    for (int j = 0; j < 4; ++j) {
      int n = n0 + j * 16 + nl;
      int h = n >> 6, d = n & 63;
#pragma unroll
      for (int r = 0; r < 8; ++r) {
        int row = m0 + r + 8 * g;
        int b = row >> 11, s = row & 2047;
        half_t val = (half_t)cc[j][r];
        if (mat == 0)
          Qh[((b * NHEADS + h) * SEQ + s) * HEADD + d] = val;
        else if (mat == 1)
          Kh[((b * NHEADS + h) * SEQ + s) * HEADD + d] = val;
        else
          Vt[((b * NHEADS + h) * HEADD + d) * SEQ + s] = val;
      }
    }
  });
}

// Output projection: AO(f16) @ Wo -> f32 out. grid (64, 8), block 256.
__global__ void __launch_bounds__(256) k_gemm_out(const half_t* __restrict__ AO,
                                                  const half_t* __restrict__ Wot,
                                                  float* __restrict__ out) {
  __shared__ __align__(16) half_t wls[64 * 256];  // 32 KB
  const int wave = threadIdx.x >> 5;
  const int lane = threadIdx.x & 31;
  const int m0 = blockIdx.x * 128 + wave * 16;
  const int n0 = blockIdx.y * 64;

  gemm_strip(AO, Wot, wls, m0, n0, [&](v8f c0, v8f c1, v8f c2, v8f c3) {
    const int g = lane >> 4;
    const int nl = lane & 15;
#pragma unroll
    for (int r = 0; r < 8; ++r) {
      float* dst = out + (size_t)(m0 + r + 8 * g) * HIDDEN + n0 + nl;
      dst[0] = c0[r];
      dst[16] = c1[r];
      dst[32] = c2[r];
      dst[48] = c3[r];
    }
  });
}

// ---------------------------------------------------------------------------
// Flash attention: grid (16, H, B), block 256; each wave owns 16 Q rows.
// K/V tiles (64 keys) staged cooperatively via double-buffered async LDS
// pipeline shared by all 8 waves; 16 WMMAs per softmax update.
// ---------------------------------------------------------------------------
static __device__ __forceinline__ void stage_kv(const half_t* Kp,
                                                const half_t* Vp, int t0,
                                                half_t* kdst, half_t* vdst) {
  const int tid = threadIdx.x;
  // K tile: rows t0..t0+63 of [t][64] -> linear 8 KB
#pragma unroll
  for (int j = 0; j < 2; ++j) {
    int c = tid + j * 256;
    copy16_g2l(Kp + t0 * HEADD + c * 8, kdst + c * 8);
  }
  // V tile: rows d=0..63 of [d][SEQ], t-slice [t0, t0+64)
#pragma unroll
  for (int j = 0; j < 2; ++j) {
    int c = tid + j * 256;
    int d = c >> 3, o = (c & 7) * 8;
    copy16_g2l(Vp + d * SEQ + t0 + o, vdst + c * 8);
  }
}

__global__ void __launch_bounds__(256) k_attn(
    const half_t* __restrict__ Qh, const half_t* __restrict__ Kh,
    const half_t* __restrict__ Vt, const half_t* __restrict__ bias_full,
    half_t* __restrict__ AO) {
  __shared__ half_t bias_s[4096];                    //  8 KB
  __shared__ __align__(16) half_t pbuf[8][16 * 64];  // 16 KB
  __shared__ __align__(16) half_t kls[2][64 * 64];   // 16 KB
  __shared__ __align__(16) half_t vls[2][64 * 64];   // 16 KB

  const int wave = threadIdx.x >> 5;
  const int lane = threadIdx.x & 31;
  const int g = lane >> 4;
  const int nl = lane & 15;
  const int h = blockIdx.y;
  const int b = blockIdx.z;
  const int s0 = blockIdx.x * 128 + wave * 16;

  const half_t* Qp = Qh + (b * NHEADS + h) * SEQ * HEADD;
  const half_t* Kp = Kh + (b * NHEADS + h) * SEQ * HEADD;
  const half_t* Vp = Vt + (b * NHEADS + h) * HEADD * SEQ;
  half_t* pw = pbuf[wave];

  for (int i = threadIdx.x; i < 4096; i += blockDim.x)
    bias_s[i] = bias_full[h * 4096 + i];

  // prime the pipeline with key block 0
  stage_kv(Kp, Vp, 0, kls[0], vls[0]);

  // Q rows held in registers for the whole pass (scale folded into Wq).
  v16h aq0 = load_a_frag(Qp, HEADD, s0, 0);
  v16h aq1 = load_a_frag(Qp, HEADD, s0, 32);

  float mrow[8], lrow[8];
#pragma unroll
  for (int r = 0; r < 8; ++r) { mrow[r] = -1e30f; lrow[r] = 0.0f; }
  v8f o0 = {}, o1 = {}, o2 = {}, o3 = {};

  for (int it = 0; it < NTBLK; ++it) {
    const int t0 = it * 64;
    __syncthreads();  // (a) everyone done reading the buffer we overwrite next
    if (it + 1 < NTBLK) {
      stage_kv(Kp, Vp, t0 + 64, kls[(it + 1) & 1], vls[(it + 1) & 1]);
      wait_async_le4();  // async in-order: <=4 pending => block `it` landed
    } else {
      wait_async_0();
    }
    __syncthreads();  // (b) block `it` visible to all waves
    const half_t* kb = kls[it & 1];
    const half_t* vb = vls[it & 1];

    // --- scores: four 16x16 tiles covering keys [t0, t0+64) ---
    v8f sc[4] = {{}, {}, {}, {}};
#pragma unroll
    for (int j = 0; j < 4; ++j) {
      v16h bk0 = load_b_frag(kb, HEADD, j * 16, 0);
      v16h bk1 = load_b_frag(kb, HEADD, j * 16, 32);
      sc[j] = wmma_f32_f16(aq0, bk0, sc[j]);
      sc[j] = wmma_f32_f16(aq1, bk1, sc[j]);
    }

    // --- bias + online softmax over 64 columns ---
    float alpha[8];
#pragma unroll
    for (int r = 0; r < 8; ++r) {
      const int srow = s0 + r + 8 * g;
      const int base = t0 + nl - srow + 2048;
      float v0 = sc[0][r] + (float)bias_s[base];
      float v1 = sc[1][r] + (float)bias_s[base + 16];
      float v2 = sc[2][r] + (float)bias_s[base + 32];
      float v3 = sc[3][r] + (float)bias_s[base + 48];
      float mx = fmaxf(fmaxf(v0, v1), fmaxf(v2, v3));
      mx = fmaxf(mx, __shfl_xor(mx, 1, 32));
      mx = fmaxf(mx, __shfl_xor(mx, 2, 32));
      mx = fmaxf(mx, __shfl_xor(mx, 4, 32));
      mx = fmaxf(mx, __shfl_xor(mx, 8, 32));
      float mnew = fmaxf(mrow[r], mx);
      float al = __expf(mrow[r] - mnew);
      mrow[r] = mnew;
      float p0 = __expf(v0 - mnew);
      float p1 = __expf(v1 - mnew);
      float p2 = __expf(v2 - mnew);
      float p3 = __expf(v3 - mnew);
      float rs = (p0 + p1) + (p2 + p3);
      rs += __shfl_xor(rs, 1, 32);
      rs += __shfl_xor(rs, 2, 32);
      rs += __shfl_xor(rs, 4, 32);
      rs += __shfl_xor(rs, 8, 32);
      lrow[r] = lrow[r] * al + rs;
      alpha[r] = al;
      // stage P (f32 D-layout) into LDS as f16 [16 s][64 t] row-major
      half_t* prow = pw + (r + 8 * g) * 64 + nl;
      prow[0] = (half_t)p0;
      prow[16] = (half_t)p1;
      prow[32] = (half_t)p2;
      prow[48] = (half_t)p3;
    }

#pragma unroll
    for (int r = 0; r < 8; ++r) {
      o0[r] *= alpha[r];
      o1[r] *= alpha[r];
      o2[r] *= alpha[r];
      o3[r] *= alpha[r];
    }

    // wave-internal cross-lane LDS RAW: wait all ds stores before reading
    asm volatile("s_wait_dscnt 0" ::: "memory");

    // --- P(16x64) @ V(64x64) from LDS ---
    v16h ap0 = load_a_frag(pw, 64, 0, 0);
    v16h ap1 = load_a_frag(pw, 64, 0, 32);
    o0 = wmma_f32_f16(ap0, load_b_frag(vb, 64, 0, 0), o0);
    o1 = wmma_f32_f16(ap0, load_b_frag(vb, 64, 16, 0), o1);
    o2 = wmma_f32_f16(ap0, load_b_frag(vb, 64, 32, 0), o2);
    o3 = wmma_f32_f16(ap0, load_b_frag(vb, 64, 48, 0), o3);
    o0 = wmma_f32_f16(ap1, load_b_frag(vb, 64, 0, 32), o0);
    o1 = wmma_f32_f16(ap1, load_b_frag(vb, 64, 16, 32), o1);
    o2 = wmma_f32_f16(ap1, load_b_frag(vb, 64, 32, 32), o2);
    o3 = wmma_f32_f16(ap1, load_b_frag(vb, 64, 48, 32), o3);
  }

  // epilogue: normalize and store head-interleaved f16 [b][s][h*64+d]
#pragma unroll
  for (int r = 0; r < 8; ++r) {
    float inv = 1.0f / lrow[r];
    int row = b * SEQ + s0 + r + 8 * g;
    half_t* dst = AO + row * HIDDEN + h * HEADD + nl;
    dst[0] = (half_t)(o0[r] * inv);
    dst[16] = (half_t)(o1[r] * inv);
    dst[32] = (half_t)(o2[r] * inv);
    dst[48] = (half_t)(o3[r] * inv);
  }
}

// ---------------------------------------------------------------------------
extern "C" void kernel_launch(void* const* d_in, const int* in_sizes, int n_in,
                              void* d_out, int out_size, void* d_ws,
                              size_t ws_size, hipStream_t stream) {
  (void)in_sizes; (void)n_in; (void)out_size; (void)ws_size;
  const float* x = (const float*)d_in[0];
  const float* Wq = (const float*)d_in[1];
  const float* Wk = (const float*)d_in[2];
  const float* Wv = (const float*)d_in[3];
  const float* Wo = (const float*)d_in[4];
  const float* bt = (const float*)d_in[5];

  char* ws = (char*)d_ws;
  size_t off = 0;
  auto alloc = [&](size_t bytes) -> void* {
    void* p = ws + off;
    off += (bytes + 255) & ~(size_t)255;
    return p;
  };
  half_t* xh = (half_t*)alloc((size_t)ROWS * HIDDEN * 2);          // 8 MB
  half_t* Wqt = (half_t*)alloc((size_t)HIDDEN * HIDDEN * 2);
  half_t* Wkt = (half_t*)alloc((size_t)HIDDEN * HIDDEN * 2);
  half_t* Wvt = (half_t*)alloc((size_t)HIDDEN * HIDDEN * 2);
  half_t* Wot = (half_t*)alloc((size_t)HIDDEN * HIDDEN * 2);
  half_t* Qh = (half_t*)alloc((size_t)ROWS * HIDDEN * 2);          // 8 MB
  half_t* Kh = (half_t*)alloc((size_t)ROWS * HIDDEN * 2);          // 8 MB
  half_t* Vt = (half_t*)alloc((size_t)ROWS * HIDDEN * 2);          // 8 MB
  half_t* AO = (half_t*)alloc((size_t)ROWS * HIDDEN * 2);          // 8 MB
  half_t* bias_full = (half_t*)alloc((size_t)NHEADS * 4096 * 2);   // 64 KB

  k_convert_x<<<4096, 256, 0, stream>>>(x, xh, ROWS * HIDDEN);
  k_convert_w<<<dim3(256, 4), 256, 0, stream>>>(Wq, Wk, Wv, Wo, Wqt, Wkt, Wvt,
                                                Wot);
  k_bias<<<16, 256, 0, stream>>>(bt, bias_full);
  k_gemm_qkv<<<dim3(64, 8, 3), 256, 0, stream>>>(xh, Wqt, Wkt, Wvt, Qh, Kh, Vt);
  k_attn<<<dim3(16, NHEADS, BATCH), 256, 0, stream>>>(Qh, Kh, Vt, bias_full,
                                                      AO);
  k_gemm_out<<<dim3(64, 8), 256, 0, stream>>>(AO, Wot, (float*)d_out);
}
